// DRMamba_73023033966774
// MI455X (gfx1250) — matrix operand order repaired and checked by reading, not compiled, not run on hardware
//
#include <hip/hip_runtime.h>
#include <hip/hip_bf16.h>

// ---- problem constants (from reference) ----
#define D_MODEL  64
#define D_STATE  16
#define D_CONVK  4
#define D_INNER  128
#define DT_RANK  4          // (64+15)//16
#define XP_ROWS  36         // DT_RANK + 2*D_STATE
#define NB       4
#define SEQ      16384      // H*W = 128*128
#define NGRP     64         // scan groups
#define LGRP     256        // SEQ/NGRP

typedef __attribute__((ext_vector_type(16))) _Float16 v16h;
typedef __attribute__((ext_vector_type(8)))  float    v8f;

// ---------------- WMMA fragment builders (contiguous k order, same in A and B) ----------------
// A (16x32 f16), row-major LDS: lane m = lane&15; K(e) = kb + kh*16 + e.
// One contiguous 32B run per lane -> ds_load_b128 pairs.
__device__ __forceinline__ v16h frag_a(const _Float16* w, int m, int ldw, int kb, int kh) {
  v16h a;
#pragma unroll
  for (int e = 0; e < 16; ++e) {
    int k = kb + e + 16 * kh;
    a[e] = w[m * ldw + k];
  }
  return a;
}
// B (32x16 f16) from K-major (transposed) activation tile act_t[col][k]:
// lane n = lane&15 at column col; K = kb + kh*16 + e  -> 16 contiguous halves
// (32B, 16B-aligned) -> two ds_load_b128, no per-half packing.
__device__ __forceinline__ v16h frag_b_t(const _Float16* act_t, int ld, int col, int kb, int kh) {
  const _Float16* p = act_t + col * ld + kb + kh * 16;
  v16h b;
#pragma unroll
  for (int e = 0; e < 16; ++e) b[e] = p[e];
  return b;
}

// ---------------- GEMM 1: xz^T = W_in(rev) @ x^T  -> xi^T, z^T ----------------
// 4 waves/block, each wave owns TWO 16-col N-chunks (A-frag reuse x2).
__global__ __launch_bounds__(128) void k_gemm_in(const float* __restrict__ x,
                                                 const float* __restrict__ W_in,
                                                 float* __restrict__ xi,
                                                 float* __restrict__ z) {
  __shared__ __align__(16) _Float16 lw[2 * D_INNER * D_MODEL];   // 256x64 (reversed W_in)
  __shared__ __align__(16) _Float16 lx[128 * 72];                // x tile, K-major, padded
  const int tid = threadIdx.x;
  const int b   = blockIdx.y;
  const int l0  = blockIdx.x * 128;

  for (int i = tid; i < 2 * D_INNER * D_MODEL; i += 128) {
    int j = i >> 6, c = i & 63;
    lw[i] = (_Float16)W_in[j * D_MODEL + (D_MODEL - 1 - c)];  // fold channel reversal
  }
  const float* xb = x + (size_t)b * D_MODEL * SEQ;
  for (int i = tid; i < D_MODEL * 128; i += 128) {
    int c = i >> 7, col = i & 127;                    // coalesced global read
    lx[col * 72 + c] = (_Float16)xb[(size_t)c * SEQ + l0 + col];
  }
  __syncthreads();

  const int wv = tid >> 5, lane = tid & 31;
  const int kh = lane >> 4, ln = lane & 15;
  const int col0 = wv * 32 + ln, col1 = col0 + 16;
  const int lA = l0 + col0, lB = l0 + col1;
  v16h b00 = frag_b_t(lx, 72, col0, 0,  kh);
  v16h b01 = frag_b_t(lx, 72, col0, 32, kh);
  v16h b10 = frag_b_t(lx, 72, col1, 0,  kh);
  v16h b11 = frag_b_t(lx, 72, col1, 32, kh);
#pragma unroll
  for (int mt = 0; mt < 16; ++mt) {
    v16h a0 = frag_a(lw, mt * 16 + ln, D_MODEL, 0, kh);
    v16h a1 = frag_a(lw, mt * 16 + ln, D_MODEL, 32, kh);
    v8f accA = {}, accB = {};
    accA = __builtin_amdgcn_wmma_f32_16x16x32_f16(false, a0, false, b00, (short)0, accA, false, false);
    accA = __builtin_amdgcn_wmma_f32_16x16x32_f16(false, a1, false, b01, (short)0, accA, false, false);
    accB = __builtin_amdgcn_wmma_f32_16x16x32_f16(false, a0, false, b10, (short)0, accB, false, false);
    accB = __builtin_amdgcn_wmma_f32_16x16x32_f16(false, a1, false, b11, (short)0, accB, false, false);
#pragma unroll
    for (int r = 0; r < 8; ++r) {
      int j = mt * 16 + kh * 8 + r;
      float* dstA = (j < D_INNER) ? (xi + ((size_t)(b * D_INNER + j)) * SEQ + lA)
                                  : (z  + ((size_t)(b * D_INNER + (j - D_INNER))) * SEQ + lA);
      float* dstB = (j < D_INNER) ? (xi + ((size_t)(b * D_INNER + j)) * SEQ + lB)
                                  : (z  + ((size_t)(b * D_INNER + (j - D_INNER))) * SEQ + lB);
      *dstA = accA[r];
      *dstB = accB[r];
    }
  }
}

// ---------------- causal depthwise conv(4) + SiLU (channel-major) ----------------
__global__ void k_conv(const float* __restrict__ xi, const float* __restrict__ cw,
                       const float* __restrict__ cb, float* __restrict__ xc, int total) {
  int idx = blockIdx.x * blockDim.x + threadIdx.x;
  if (idx >= total) return;
  int l  = idx & (SEQ - 1);
  int bd = idx >> 14;                  // SEQ == 1<<14
  int d  = bd & (D_INNER - 1);
  const float* row = xi + (size_t)bd * SEQ;
  __builtin_prefetch(row + l + 256, 0, 0);
  float v = cb[d];
#pragma unroll
  for (int k = 0; k < D_CONVK; ++k) {
    int ll = l - (D_CONVK - 1) + k;
    if (ll >= 0) v += cw[d * D_CONVK + k] * row[ll];
  }
  xc[idx] = v / (1.0f + __expf(-v));   // silu
}

// ---------------- GEMM 2: dbl^T = W_xproj @ xc^T ----------------
__global__ __launch_bounds__(128) void k_gemm_xproj(const float* __restrict__ xc,
                                                    const float* __restrict__ W_xproj,
                                                    float* __restrict__ dbl) {
  __shared__ __align__(16) _Float16 lw[48 * D_INNER];   // 36 rows padded to 48
  __shared__ __align__(16) _Float16 la[128 * 136];      // act tile, K-major, padded
  const int tid = threadIdx.x;
  const int b   = blockIdx.y;
  const int l0  = blockIdx.x * 128;

  for (int i = tid; i < 48 * D_INNER; i += 128) {
    int r = i >> 7, c = i & 127;
    lw[i] = (r < XP_ROWS) ? (_Float16)W_xproj[r * D_INNER + c] : (_Float16)0.0f;
  }
  const float* src = xc + (size_t)b * D_INNER * SEQ;
  for (int i = tid; i < D_INNER * 128; i += 128) {
    int d = i >> 7, col = i & 127;
    la[col * 136 + d] = (_Float16)src[(size_t)d * SEQ + l0 + col];
  }
  __syncthreads();

  const int wv = tid >> 5, lane = tid & 31;
  const int kh = lane >> 4, ln = lane & 15;
  const int col0 = wv * 32 + ln, col1 = col0 + 16;
  const int lA = l0 + col0, lB = l0 + col1;
  v16h bf0[4], bf1[4];
#pragma unroll
  for (int ks = 0; ks < 4; ++ks) {
    bf0[ks] = frag_b_t(la, 136, col0, ks * 32, kh);
    bf1[ks] = frag_b_t(la, 136, col1, ks * 32, kh);
  }
#pragma unroll
  for (int mt = 0; mt < 3; ++mt) {
    v8f accA = {}, accB = {};
#pragma unroll
    for (int ks = 0; ks < 4; ++ks) {
      v16h a = frag_a(lw, mt * 16 + ln, D_INNER, ks * 32, kh);
      accA = __builtin_amdgcn_wmma_f32_16x16x32_f16(false, a, false, bf0[ks], (short)0, accA, false, false);
      accB = __builtin_amdgcn_wmma_f32_16x16x32_f16(false, a, false, bf1[ks], (short)0, accB, false, false);
    }
#pragma unroll
    for (int r = 0; r < 8; ++r) {
      int j = mt * 16 + kh * 8 + r;
      if (j < XP_ROWS) {
        dbl[((size_t)(b * XP_ROWS + j)) * SEQ + lA] = accA[r];
        dbl[((size_t)(b * XP_ROWS + j)) * SEQ + lB] = accB[r];
      }
    }
  }
}

// ---------------- chunk-parallel selective scan ----------------
__device__ __forceinline__ float softplus_f(float x) {
  return (x > 20.f) ? x : log1pf(__expf(x));
}

__global__ __launch_bounds__(32) void k_scan_a(const float* __restrict__ dbl,
                                               const float* __restrict__ xc,
                                               const float* __restrict__ W_dt,
                                               const float* __restrict__ b_dt,
                                               const float* __restrict__ A_log,
                                               float* __restrict__ Pg, float* __restrict__ Sg) {
  const int g = blockIdx.x, dp = blockIdx.y, b = blockIdx.z;
  const int lane = threadIdx.x;
  const int n = lane & 15, d = dp * 2 + (lane >> 4);
  const float A  = -__expf(A_log[d * D_STATE + n]);
  const float w0 = W_dt[d * DT_RANK + 0], w1 = W_dt[d * DT_RANK + 1];
  const float w2 = W_dt[d * DT_RANK + 2], w3 = W_dt[d * DT_RANK + 3];
  const float bd = b_dt[d];
  const float* r0   = dbl + ((size_t)b * XP_ROWS) * SEQ;
  const float* r1   = r0 + SEQ;
  const float* r2   = r0 + 2 * (size_t)SEQ;
  const float* r3   = r0 + 3 * (size_t)SEQ;
  const float* Brow = dbl + ((size_t)b * XP_ROWS + DT_RANK + n) * SEQ;
  const float* xrow = xc  + ((size_t)(b * D_INNER + d)) * SEQ;
  float P = 1.f, S = 0.f;
  const int l0 = g * LGRP;
  for (int t = 0; t < LGRP; ++t) {
    int l = l0 + t;
    if ((t & 63) == 0) __builtin_prefetch(xrow + l + 64, 0, 0);
    float dt = softplus_f(bd + w0 * r0[l] + w1 * r1[l] + w2 * r2[l] + w3 * r3[l]);
    float a  = __expf(dt * A);
    P *= a;
    S  = a * S + dt * xrow[l] * Brow[l];
  }
  size_t base = (((size_t)(b * D_INNER + d)) * D_STATE + n) * NGRP + g;
  Pg[base] = P; Sg[base] = S;
}

__global__ __launch_bounds__(32) void k_scan_b(const float* __restrict__ Pg,
                                               const float* __restrict__ Sg,
                                               float* __restrict__ Cg) {
  const int dp = blockIdx.x, b = blockIdx.y;
  const int lane = threadIdx.x;
  const int n = lane & 15, d = dp * 2 + (lane >> 4);
  size_t base = (((size_t)(b * D_INNER + d)) * D_STATE + n) * NGRP;
  float c = 0.f;
  for (int g = 0; g < NGRP; ++g) {
    Cg[base + g] = c;                       // carry-in for group g
    c = Pg[base + g] * c + Sg[base + g];
  }
}

__global__ __launch_bounds__(32) void k_scan_c(const float* __restrict__ dbl,
                                               const float* __restrict__ xc,
                                               const float* __restrict__ zT,
                                               const float* __restrict__ W_dt,
                                               const float* __restrict__ b_dt,
                                               const float* __restrict__ A_log,
                                               const float* __restrict__ D_skip,
                                               const float* __restrict__ Cg,
                                               float* __restrict__ yT) {
  const int g = blockIdx.x, dp = blockIdx.y, b = blockIdx.z;
  const int lane = threadIdx.x;
  const int n = lane & 15, d = dp * 2 + (lane >> 4);
  const float A  = -__expf(A_log[d * D_STATE + n]);
  const float w0 = W_dt[d * DT_RANK + 0], w1 = W_dt[d * DT_RANK + 1];
  const float w2 = W_dt[d * DT_RANK + 2], w3 = W_dt[d * DT_RANK + 3];
  const float bd = b_dt[d];
  const float Dsk = D_skip[d];
  const float* r0   = dbl + ((size_t)b * XP_ROWS) * SEQ;
  const float* r1   = r0 + SEQ;
  const float* r2   = r0 + 2 * (size_t)SEQ;
  const float* r3   = r0 + 3 * (size_t)SEQ;
  const float* Brow = dbl + ((size_t)b * XP_ROWS + DT_RANK + n) * SEQ;
  const float* Crow = dbl + ((size_t)b * XP_ROWS + DT_RANK + D_STATE + n) * SEQ;
  const float* xrow = xc + ((size_t)(b * D_INNER + d)) * SEQ;
  const float* zrow = zT + ((size_t)(b * D_INNER + d)) * SEQ;
  float*       yrow = yT + ((size_t)(b * D_INNER + d)) * SEQ;
  float h = Cg[(((size_t)(b * D_INNER + d)) * D_STATE + n) * NGRP + g];
  const int l0 = g * LGRP;
  for (int t = 0; t < LGRP; ++t) {
    int l = l0 + t;
    float dt = softplus_f(bd + w0 * r0[l] + w1 * r1[l] + w2 * r2[l] + w3 * r3[l]);
    float a  = __expf(dt * A);
    float xt = xrow[l];
    h = a * h + dt * xt * Brow[l];
    float p = h * Crow[l];
    p += __shfl_xor(p, 1, 32);              // reduce within 16-lane state group
    p += __shfl_xor(p, 2, 32);
    p += __shfl_xor(p, 4, 32);
    p += __shfl_xor(p, 8, 32);
    if (n == 0) {
      float y  = p + xt * Dsk;
      float zv = zrow[l];
      y *= zv / (1.f + __expf(-zv));        // * silu(z)
      yrow[l] = y;
    }
  }
}

// ---------------- GEMM 3: out^T = W_out @ y^T (rows written reversed) ----------------
__global__ __launch_bounds__(128) void k_gemm_out(const float* __restrict__ yT,
                                                  const float* __restrict__ W_out,
                                                  float* __restrict__ out) {
  __shared__ __align__(16) _Float16 lw[D_MODEL * D_INNER];  // 64x128
  __shared__ __align__(16) _Float16 la[128 * 136];
  const int tid = threadIdx.x;
  const int b   = blockIdx.y;
  const int l0  = blockIdx.x * 128;

  for (int i = tid; i < D_MODEL * D_INNER; i += 128) lw[i] = (_Float16)W_out[i];
  const float* src = yT + (size_t)b * D_INNER * SEQ;
  for (int i = tid; i < D_INNER * 128; i += 128) {
    int d = i >> 7, col = i & 127;
    la[col * 136 + d] = (_Float16)src[(size_t)d * SEQ + l0 + col];
  }
  __syncthreads();

  const int wv = tid >> 5, lane = tid & 31;
  const int kh = lane >> 4, ln = lane & 15;
  const int col0 = wv * 32 + ln, col1 = col0 + 16;
  const int lA = l0 + col0, lB = l0 + col1;
  v16h bf0[4], bf1[4];
#pragma unroll
  for (int ks = 0; ks < 4; ++ks) {
    bf0[ks] = frag_b_t(la, 136, col0, ks * 32, kh);
    bf1[ks] = frag_b_t(la, 136, col1, ks * 32, kh);
  }
#pragma unroll
  for (int mt = 0; mt < 4; ++mt) {
    v8f accA = {}, accB = {};
#pragma unroll
    for (int ks = 0; ks < 4; ++ks) {
      v16h a = frag_a(lw, mt * 16 + ln, D_INNER, ks * 32, kh);
      accA = __builtin_amdgcn_wmma_f32_16x16x32_f16(false, a, false, bf0[ks], (short)0, accA, false, false);
      accB = __builtin_amdgcn_wmma_f32_16x16x32_f16(false, a, false, bf1[ks], (short)0, accB, false, false);
    }
#pragma unroll
    for (int r = 0; r < 8; ++r) {
      int j = mt * 16 + kh * 8 + r;                       // model output channel
      out[((size_t)(b * D_MODEL + (D_MODEL - 1 - j))) * SEQ + lA] = accA[r]; // fold reversal
      out[((size_t)(b * D_MODEL + (D_MODEL - 1 - j))) * SEQ + lB] = accB[r];
    }
  }
}

// ---------------- host launcher ----------------
extern "C" void kernel_launch(void* const* d_in, const int* in_sizes, int n_in,
                              void* d_out, int out_size, void* d_ws, size_t ws_size,
                              hipStream_t stream) {
  const float* x      = (const float*)d_in[0];
  const float* W_in   = (const float*)d_in[1];
  const float* conv_w = (const float*)d_in[2];
  const float* conv_b = (const float*)d_in[3];
  const float* W_xprj = (const float*)d_in[4];
  const float* W_dt   = (const float*)d_in[5];
  const float* b_dt   = (const float*)d_in[6];
  const float* A_log  = (const float*)d_in[7];
  const float* D_skip = (const float*)d_in[8];
  const float* W_out  = (const float*)d_in[9];
  float* out = (float*)d_out;

  char* ws = (char*)d_ws;
  const size_t szCh  = (size_t)NB * D_INNER * SEQ * sizeof(float);   // 32 MB
  const size_t szDbl = (size_t)NB * XP_ROWS * SEQ * sizeof(float);   // 9 MB
  const size_t nPS   = (size_t)NB * D_INNER * D_STATE * NGRP;        // 512K elems
  float* xi  = (float*)(ws);
  float* z   = (float*)(ws + szCh);
  float* xc  = (float*)(ws + 2 * szCh);
  float* dbl = (float*)(ws + 3 * szCh);
  float* Pg  = (float*)(ws + 3 * szCh + szDbl);
  float* Sg  = Pg + nPS;
  float* Cg  = Sg + nPS;
  float* yT  = xi;   // xi is dead after the conv -> reuse for y^T

  dim3 gg(SEQ / 128, NB);
  k_gemm_in<<<gg, 128, 0, stream>>>(x, W_in, xi, z);
  int total = NB * D_INNER * SEQ;
  k_conv<<<(total + 255) / 256, 256, 0, stream>>>(xi, conv_w, conv_b, xc, total);
  k_gemm_xproj<<<gg, 128, 0, stream>>>(xc, W_xprj, dbl);
  k_scan_a<<<dim3(NGRP, D_INNER / 2, NB), 32, 0, stream>>>(dbl, xc, W_dt, b_dt, A_log, Pg, Sg);
  k_scan_b<<<dim3(D_INNER / 2, NB), 32, 0, stream>>>(Pg, Sg, Cg);
  k_scan_c<<<dim3(NGRP, D_INNER / 2, NB), 32, 0, stream>>>(dbl, xc, z, W_dt, b_dt, A_log,
                                                           D_skip, Cg, yT);
  k_gemm_out<<<gg, 128, 0, stream>>>(yT, W_out, out);
}
